// NMI_33509334843740
// MI455X (gfx1250) — matrix operand order repaired and checked by reading, not compiled, not run on hardware
//
#include <hip/hip_runtime.h>
#include <hip/hip_bf16.h>
#include <math.h>

typedef __attribute__((ext_vector_type(16))) _Float16 v16h;
typedef __attribute__((ext_vector_type(8)))  float    v8f;

#define NBINS   64
#define EPSV    1e-10f
#define INVNORM2 15.915494309189534f   // 1/(2*pi*sigma^2) = (1/(sqrt(2pi)*sigma))^2
// C2 = -1/(2*sigma^2) * log2(e) = -50 * 1.4426950408889634
#define C2      (-72.13475204444817f)

// ---------------------------------------------------------------------------
// Kernel 0: initialize workspace (min/max slots as uint bit patterns; data >= 0
// so unsigned ordering == float ordering), zero the joint-histogram accumulator.
// ---------------------------------------------------------------------------
__global__ __launch_bounds__(256) void nmi_init(unsigned* mm, float* pxy) {
    int t = threadIdx.x;
    if (t == 0) {
        mm[0] = 0x7F800000u; mm[1] = 0u;   // fixed  min=+inf, max=0
        mm[2] = 0x7F800000u; mm[3] = 0u;   // warped min=+inf, max=0
    }
    for (int i = t; i < 2 * NBINS * NBINS; i += 256) pxy[i] = 0.0f;
}

// ---------------------------------------------------------------------------
// Kernel 1: global min/max of both volumes (block LDS tree + uint atomics).
// ---------------------------------------------------------------------------
__global__ __launch_bounds__(256) void nmi_minmax(const float* __restrict__ fx,
                                                  const float* __restrict__ wp,
                                                  unsigned* mm, int n) {
    __shared__ float sflo[256], sfhi[256], swlo[256], swhi[256];
    int t = threadIdx.x;
    int tid = blockIdx.x * 256 + t;
    int stride = gridDim.x * 256;
    float flo = __uint_as_float(0x7F800000u), fhi = 0.0f;
    float wlo = flo, whi = 0.0f;
    for (int i = tid; i < n; i += stride) {
        float f = fx[i]; flo = fminf(flo, f); fhi = fmaxf(fhi, f);
        float w = wp[i]; wlo = fminf(wlo, w); whi = fmaxf(whi, w);
    }
    sflo[t] = flo; sfhi[t] = fhi; swlo[t] = wlo; swhi[t] = whi;
    __syncthreads();
    for (int off = 128; off > 0; off >>= 1) {
        if (t < off) {
            sflo[t] = fminf(sflo[t], sflo[t + off]);
            sfhi[t] = fmaxf(sfhi[t], sfhi[t + off]);
            swlo[t] = fminf(swlo[t], swlo[t + off]);
            swhi[t] = fmaxf(swhi[t], swhi[t + off]);
        }
        __syncthreads();
    }
    if (t == 0) {
        atomicMin(&mm[0], __float_as_uint(sflo[0]));
        atomicMax(&mm[1], __float_as_uint(sfhi[0]));
        atomicMin(&mm[2], __float_as_uint(swlo[0]));
        atomicMax(&mm[3], __float_as_uint(swhi[0]));
    }
}

// ---------------------------------------------------------------------------
// Kernel 2: joint Parzen histogram via WMMA.
// p_XY[b] (64x64) = sum_n w1[b,:,n] * w2[b,:,n]^T  — a 64x64xK GEMM, K = N.
// Each wave handles 32-sample K-chunks. The K->sample mapping is chosen so the
// per-lane fragment slots are contiguous memory runs, so samples are fetched
// directly with float4 loads (no cross-lane shuffles):
//   A (16x32 f16 layout): lane needs [hi*8..+7] and [16+hi*8..+7]
//   B (32x16 f16 layout): lane needs [hi*8..+7] and [16+hi*8..+7]
// Windows use the expanded quadratic C2*(x-b)^2 = u(x) + k1(b)*x + k0(b), so
// each window costs fmac + add + v_exp_f32 (exp2 base). 16 v_wmma per chunk
// with f32 register accumulators; LDS-reduction epilogue, one global atomic
// pass per block.
// ---------------------------------------------------------------------------
__global__ __launch_bounds__(256)
void nmi_pxy(const float* __restrict__ fx, const float* __restrict__ wp,
             const unsigned* __restrict__ mm, float* __restrict__ pxy, int N) {
    __shared__ float sAcc[NBINS * NBINS];

    const int lane = threadIdx.x & 31;
    const int wv   = threadIdx.x >> 5;
    const int wgid = blockIdx.x * 8 + wv;   // wave id within batch
    const int nw   = gridDim.x * 8;         // waves per batch
    const int b    = blockIdx.y;
    const int hi   = lane >> 4;             // lane half (ISA fragment layouts)
    const int mr   = lane & 15;             // row (A) / col (B,C,D) within tile

    for (int i = threadIdx.x; i < NBINS * NBINS; i += 256) sAcc[i] = 0.0f;

    const float fmin = __uint_as_float(mm[0]);
    const float fmax = __uint_as_float(mm[1]);
    const float wmin = __uint_as_float(mm[2]);
    const float wmax = __uint_as_float(mm[3]);
    const float bs1 = (fmax - fmin) * (1.0f / 63.0f);
    const float bs2 = (wmax - wmin) * (1.0f / 63.0f);

    // Per-fragment loop-invariant quadratic coefficients:
    //   e(x) = C2*x^2 + k1*x + k0,  k1 = -2*C2*b,  k0 = C2*b^2
    float k1A[4], k0A[4], k1B[4], k0B[4];
#pragma unroll
    for (int i = 0; i < 4; ++i) {
        float bA = fmin + (float)(i * 16 + mr) * bs1;
        float bB = wmin + (float)(i * 16 + mr) * bs2;
        k1A[i] = -2.0f * C2 * bA;  k0A[i] = C2 * bA * bA;
        k1B[i] = -2.0f * C2 * bB;  k0B[i] = C2 * bB * bB;
    }

    v8f acc[4][4];
#pragma unroll
    for (int i = 0; i < 4; ++i)
#pragma unroll
        for (int j = 0; j < 4; ++j) {
            v8f z = {0.f, 0.f, 0.f, 0.f, 0.f, 0.f, 0.f, 0.f};
            acc[i][j] = z;
        }

    const float* x1 = fx + b * N;
    const float* x2 = wp + b * N;
    const int nchunks = N >> 5;             // 32 samples per chunk

    __syncthreads();                        // sAcc zeroed before use

    for (int c = wgid; c < nchunks; c += nw) {
        const float* cb1 = x1 + c * 32;
        const float* cb2 = x2 + c * 32;

        // Direct vector loads of this lane's fragment K-slots (contiguous runs).
        float4 a0 = ((const float4*)(cb1 + hi * 8))[0];        // A slots 0..3
        float4 a1 = ((const float4*)(cb1 + hi * 8))[1];        // A slots 4..7
        float4 a2 = ((const float4*)(cb1 + 16 + hi * 8))[0];   // A slots 8..11
        float4 a3 = ((const float4*)(cb1 + 16 + hi * 8))[1];   // A slots 12..15
        float4 b0 = ((const float4*)(cb2 + hi * 8))[0];        // B slots 0..3
        float4 b1 = ((const float4*)(cb2 + hi * 8))[1];        // B slots 4..7
        float4 b2 = ((const float4*)(cb2 + 16 + hi * 8))[0];   // B slots 8..11
        float4 b3 = ((const float4*)(cb2 + 16 + hi * 8))[1];   // B slots 12..15

        float xs1[16] = {a0.x, a0.y, a0.z, a0.w, a1.x, a1.y, a1.z, a1.w,
                         a2.x, a2.y, a2.z, a2.w, a3.x, a3.y, a3.z, a3.w};
        float xs2[16] = {b0.x, b0.y, b0.z, b0.w, b1.x, b1.y, b1.z, b1.w,
                         b2.x, b2.y, b2.z, b2.w, b3.x, b3.y, b3.z, b3.w};

        float u1[16], u2[16];               // u = C2 * x^2, shared by 4 frags
#pragma unroll
        for (int t = 0; t < 16; ++t) {
            u1[t] = (C2 * xs1[t]) * xs1[t];
            u2[t] = (C2 * xs2[t]) * xs2[t];
        }

        v16h bf[4];
#pragma unroll
        for (int j = 0; j < 4; ++j) {
#pragma unroll
            for (int t = 0; t < 16; ++t) {
                float e = fmaf(k1B[j], xs2[t], u2[t]) + k0B[j];
                bf[j][t] = (_Float16)__builtin_amdgcn_exp2f(e);
            }
        }
#pragma unroll
        for (int i = 0; i < 4; ++i) {
            v16h af;
#pragma unroll
            for (int t = 0; t < 16; ++t) {
                float e = fmaf(k1A[i], xs1[t], u1[t]) + k0A[i];
                af[t] = (_Float16)__builtin_amdgcn_exp2f(e);
            }
#pragma unroll
            for (int j = 0; j < 4; ++j) {
                acc[i][j] = __builtin_amdgcn_wmma_f32_16x16x32_f16(
                    false, af, false, bf[j], (short)0, acc[i][j], false, false);
            }
        }
    }

    // Wave epilogue: C/D layout — lane col = lane%16, row = vgpr + 8*(lane>=16)
    // Accumulate into block-shared histogram (ds_add_f32), then one global pass.
#pragma unroll
    for (int i = 0; i < 4; ++i)
#pragma unroll
        for (int j = 0; j < 4; ++j)
#pragma unroll
            for (int r = 0; r < 8; ++r) {
                int row = i * 16 + r + 8 * hi;
                int col = j * 16 + mr;
                atomicAdd(&sAcc[row * 64 + col], acc[i][j][r]);
            }
    __syncthreads();

    float* p = pxy + b * NBINS * NBINS;
    for (int i = threadIdx.x; i < NBINS * NBINS; i += 256)
        atomicAdd(&p[i], sAcc[i]);
}

// ---------------------------------------------------------------------------
// Kernel 3: finalize — normalize joint histogram, marginals, MI, entropies,
// NMI per batch, write scalar -mean(nmi). One block of 64 threads (row each).
// ---------------------------------------------------------------------------
__global__ __launch_bounds__(64)
void nmi_finalize(const float* __restrict__ pxy, float* __restrict__ out) {
    __shared__ float sP[NBINS * NBINS];
    __shared__ float red[NBINS], sX[NBINS], sY[NBINS], eX[NBINS], eY[NBINS];
    __shared__ float tot;
    int t = threadIdx.x;                    // 0..63, owns row t
    float nmi_sum = 0.0f;
    for (int b = 0; b < 2; ++b) {
        float rowsum = 0.0f;
        for (int j = 0; j < NBINS; ++j) {
            float v = pxy[b * NBINS * NBINS + t * NBINS + j] * INVNORM2 + EPSV;
            sP[t * NBINS + j] = v;
            rowsum += v;
        }
        red[t] = rowsum;
        __syncthreads();
        if (t == 0) {
            float s = 0.0f;
            for (int i = 0; i < NBINS; ++i) s += red[i];
            tot = s;
        }
        __syncthreads();
        float inv_tot = 1.0f / tot;
        sY[t] = rowsum * inv_tot;           // p_Y[i]: sum over j
        float colsum = 0.0f;
        for (int i = 0; i < NBINS; ++i) colsum += sP[i * NBINS + t];
        sX[t] = colsum * inv_tot;           // p_X[j]: sum over i
        __syncthreads();
        float pYt = fmaxf(sY[t], EPSV);
        float pXt = fmaxf(sX[t], EPSV);
        float Ipart = 0.0f;
        for (int j = 0; j < NBINS; ++j) {
            float q   = sP[t * NBINS + j] * inv_tot;
            float pXj = fmaxf(sX[j], EPSV);
            Ipart += q * __logf(q / (pXj * pYt));
        }
        red[t] = Ipart;
        eX[t] = pXt * __logf(pXt);
        eY[t] = pYt * __logf(pYt);
        __syncthreads();
        if (t == 0) {
            float I = 0.0f, ex = 0.0f, ey = 0.0f;
            for (int i = 0; i < NBINS; ++i) { I += red[i]; ex += eX[i]; ey += eY[i]; }
            nmi_sum += -2.0f * I / (ex + ey + EPSV);
        }
        __syncthreads();
    }
    if (t == 0) out[0] = -(nmi_sum * 0.5f);
}

// ---------------------------------------------------------------------------
extern "C" void kernel_launch(void* const* d_in, const int* in_sizes, int n_in,
                              void* d_out, int out_size, void* d_ws, size_t ws_size,
                              hipStream_t stream) {
    const float* fx = (const float*)d_in[0];   // fixed  (2,1,64,64,64) f32
    const float* wp = (const float*)d_in[1];   // warped (2,1,64,64,64) f32
    float* out = (float*)d_out;                // scalar f32
    unsigned* mm = (unsigned*)d_ws;            // [4] min/max bit patterns
    float* pxy = (float*)d_ws + 4;             // [2][64][64] f32 accumulator

    const int ntot = in_sizes[0];              // B*N = 524288
    const int N = ntot / 2;                    // per-batch voxel count

    nmi_init<<<1, 256, 0, stream>>>(mm, pxy);
    nmi_minmax<<<256, 256, 0, stream>>>(fx, wp, mm, ntot);
    dim3 grid(128, 2);                         // 1024 waves/batch, 8 chunks each
    nmi_pxy<<<grid, 256, 0, stream>>>(fx, wp, mm, pxy, N);
    nmi_finalize<<<1, 64, 0, stream>>>(pxy, out);
}
